// GCNEncoder_11029476016726
// MI455X (gfx1250) — matrix operand order repaired and checked by reading, not compiled, not run on hardware
//
#include <hip/hip_runtime.h>

typedef __attribute__((ext_vector_type(16))) _Float16 v16h;
typedef __attribute__((ext_vector_type(8)))  float    v8f;

#define IN_C 128
#define HID  128
#define OUT_C 64

// ---------------- conversion / prep kernels ----------------

__global__ void f32_to_f16_kernel(const float* __restrict__ in,
                                  _Float16* __restrict__ out, long long n) {
  long long i = (long long)blockIdx.x * blockDim.x + threadIdx.x;
  if (i < n) out[i] = (_Float16)in[i];
}

// Wt[n][k] = (f16) W[k][n]; K is always 128 here (row stride of Wt = 128)
__global__ void transpose_to_f16_kernel(const float* __restrict__ W,
                                        _Float16* __restrict__ Wt,
                                        int K, int Ncol) {
  int t = blockIdx.x * blockDim.x + threadIdx.x;
  if (t >= K * Ncol) return;
  int k = t / Ncol;
  int n = t % Ncol;
  Wt[(size_t)n * K + k] = (_Float16)W[t];
}

__global__ void deg_count_kernel(const long long* __restrict__ ei,
                                 float* __restrict__ deg, int E) {
  int e = blockIdx.x * blockDim.x + threadIdx.x;
  if (e < E) atomicAdd(&deg[ei[(long long)E + e]], 1.0f);
}

__global__ void make_dinv_kernel(const float* __restrict__ deg,
                                 float* __restrict__ dinv, int n) {
  int i = blockIdx.x * blockDim.x + threadIdx.x;
  if (i < n) dinv[i] = rsqrtf(deg[i] + 1.0f);  // +1 for self loop; deg>=1 always
}

// int64 edge list -> int32 src/dst + per-edge norm weight (reused by both layers)
__global__ void edge_prep_kernel(const long long* __restrict__ ei,
                                 const float* __restrict__ dinv,
                                 int* __restrict__ src32, int* __restrict__ dst32,
                                 float* __restrict__ wE, int E) {
  int e = blockIdx.x * blockDim.x + threadIdx.x;
  if (e >= E) return;
  const int s = (int)ei[e];
  const int d = (int)ei[(long long)E + e];
  src32[e] = s;
  dst32[e] = d;
  wE[e] = dinv[s] * dinv[d];
}

// ---------------- WMMA GEMM: C[M,NCOL] = A[M,128] * Bt[NCOL,128]^T ----------------
// A: f16 row-major [M][128].  Bt: f16 row-major [NCOL][128] (i.e. B transposed).
// One wave -> one 16x16 f32 tile; block = 4 waves covering 64 rows.

template <int NCOL>
__global__ void wmma_gemm_f16_kernel(const _Float16* __restrict__ A,
                                     const _Float16* __restrict__ Bt,
                                     float* __restrict__ C, int M) {
  const int lane    = threadIdx.x & 31;
  const int wave    = threadIdx.x >> 5;
  const int tileM   = blockIdx.x * 4 + wave;
  const int row0    = tileM * 16;
  if (row0 >= M) return;                 // uniform per wave
  const int col0    = blockIdx.y * 16;
  const int halfSel = lane >> 4;         // 0: lanes 0-15, 1: lanes 16-31
  const int lmod    = lane & 15;

  v8f acc = {0.f, 0.f, 0.f, 0.f, 0.f, 0.f, 0.f, 0.f};

  int arow = row0 + lmod;
  if (arow >= M) arow = M - 1;           // clamp: row r of C depends only on row r of A
  const _Float16* Aptr = A  + (size_t)arow * 128;
  const _Float16* Bptr = Bt + (size_t)(col0 + lmod) * 128;

#pragma unroll
  for (int kk = 0; kk < 128; kk += 32) {
    union { v16h v; unsigned u[8]; } af, bf;
#pragma unroll
    for (int r = 0; r < 8; ++r) {
      // A 16x32 f16 fragment layout (ISA 7.12.2): VGPR r, lane L holds packed pair at
      // K = (r>=4?16:0) + (L>=16?8:0) + (r&3)*2
      const int ak = kk + ((r & 4) << 2) + (halfSel << 3) + ((r & 3) << 1);
      af.u[r] = *(const unsigned*)(Aptr + ak);
      // B 32x16 fragment: lane%16 = column, VGPR r holds pair at K = (r>=4?16:0) + (L>=16?8:0) + (r&3)*2
      const int bk = kk + ((r & 4) << 2) + (halfSel << 3) + ((r & 3) << 1);
      bf.u[r] = *(const unsigned*)(Bptr + bk);
    }
    acc = __builtin_amdgcn_wmma_f32_16x16x32_f16(false, af.v, false, bf.v,
                                                 (short)0, acc, false, false);
  }

  // C/D layout: lane L, VGPR r -> N = L%16, M = (L/16)*8 + r
  const int rbase = row0 + (halfSel << 3);
  float* Cp = C + (size_t)rbase * NCOL + col0 + lmod;
  if (row0 + 16 <= M) {
    // full tile: wave-uniform fast path, unguarded stores with immediate offsets
#pragma unroll
    for (int r = 0; r < 8; ++r) Cp[r * NCOL] = acc[r];
  } else {
#pragma unroll
    for (int r = 0; r < 8; ++r) {
      if (rbase + r < M) Cp[r * NCOL] = acc[r];
    }
  }
}

// ---------------- edge gather-scale-scatter (L2-resident atomics) ----------------

template <int C>
__global__ void edge_scatter_kernel(const float* __restrict__ h,
                                    const int* __restrict__ src32,
                                    const int* __restrict__ dst32,
                                    const float* __restrict__ wE,
                                    float* __restrict__ agg, int E) {
  constexpr int TPE = C / 4;                     // threads per edge (float4 each)
  long long gid = (long long)blockIdx.x * blockDim.x + threadIdx.x;
  long long e  = gid / TPE;
  int c4       = (int)(gid % TPE);
  if (e >= E) return;
  const int   s = src32[e];
  const int   d = dst32[e];
  const float w = wE[e];
  const float4 v = *(const float4*)(h + (size_t)s * C + c4 * 4);
  float* out = agg + (size_t)d * C + (size_t)c4 * 4;
  atomicAdd(out + 0, v.x * w);
  atomicAdd(out + 1, v.y * w);
  atomicAdd(out + 2, v.z * w);
  atomicAdd(out + 3, v.w * w);
}

// ---------------- epilogues: self-loop + bias (+ReLU -> f16 for layer 1) ----------------

__global__ void finalize_relu_f16_kernel(const float* __restrict__ agg,
                                         const float* __restrict__ h,
                                         const float* __restrict__ dinv,
                                         const float* __restrict__ bias,
                                         _Float16* __restrict__ z,
                                         int n, int C) {
  int i = blockIdx.x * blockDim.x + threadIdx.x;
  if (i >= n) return;
  const int node = i / C, c = i % C;
  const float di = dinv[node];
  const float v = agg[i] + h[i] * di * di + bias[c];   // self-loop norm = 1/deg
  z[i] = (_Float16)fmaxf(v, 0.0f);
}

__global__ void finalize_out_kernel(float* __restrict__ out,
                                    const float* __restrict__ h2,
                                    const float* __restrict__ dinv,
                                    const float* __restrict__ bias,
                                    int n, int C) {
  int i = blockIdx.x * blockDim.x + threadIdx.x;
  if (i >= n) return;
  const int node = i / C, c = i % C;
  const float di = dinv[node];
  out[i] = out[i] + h2[i] * di * di + bias[c];
}

// ---------------- launcher ----------------

extern "C" void kernel_launch(void* const* d_in, const int* in_sizes, int n_in,
                              void* d_out, int out_size, void* d_ws, size_t ws_size,
                              hipStream_t stream) {
  const float*     x  = (const float*)d_in[0];
  const long long* ei = (const long long*)d_in[1];
  const float*     W1 = (const float*)d_in[2];
  const float*     b1 = (const float*)d_in[3];
  const float*     W2 = (const float*)d_in[4];
  const float*     b2 = (const float*)d_in[5];
  float* out = (float*)d_out;

  const int Nn = in_sizes[0] / IN_C;   // 50000
  const int E  = in_sizes[1] / 2;      // 800000

  // carve workspace
  char* ws = (char*)d_ws;
  size_t off = 0;
  auto carve = [&](size_t bytes) -> void* {
    void* p = ws + off;
    off = (off + bytes + 255) & ~(size_t)255;
    return p;
  };
  _Float16* x16   = (_Float16*)carve((size_t)Nn * IN_C * 2);
  _Float16* w1t   = (_Float16*)carve((size_t)HID * IN_C * 2);
  _Float16* w2t   = (_Float16*)carve((size_t)OUT_C * HID * 2);
  float*    h1    = (float*)carve((size_t)Nn * HID * 4);
  float*    agg1  = (float*)carve((size_t)Nn * HID * 4);
  _Float16* z16   = (_Float16*)carve((size_t)Nn * HID * 2);
  float*    h2    = (float*)carve((size_t)Nn * OUT_C * 4);
  float*    deg   = (float*)carve((size_t)Nn * 4);
  float*    dinv  = (float*)carve((size_t)Nn * 4);
  int*      src32 = (int*)carve((size_t)E * 4);
  int*      dst32 = (int*)carve((size_t)E * 4);
  float*    wE    = (float*)carve((size_t)E * 4);

  // zero accumulators (graph-capturable memsets)
  hipMemsetAsync(deg, 0, (size_t)Nn * 4, stream);
  hipMemsetAsync(agg1, 0, (size_t)Nn * HID * 4, stream);
  hipMemsetAsync(out, 0, (size_t)out_size * 4, stream);

  // convert / transpose weights
  {
    long long n = (long long)Nn * IN_C;
    f32_to_f16_kernel<<<(unsigned)((n + 255) / 256), 256, 0, stream>>>(x, x16, n);
  }
  transpose_to_f16_kernel<<<(IN_C * HID + 255) / 256, 256, 0, stream>>>(W1, w1t, IN_C, HID);
  transpose_to_f16_kernel<<<(HID * OUT_C + 255) / 256, 256, 0, stream>>>(W2, w2t, HID, OUT_C);

  // normalization + edge metadata (int32 src/dst + per-edge weight)
  deg_count_kernel<<<(E + 255) / 256, 256, 0, stream>>>(ei, deg, E);
  make_dinv_kernel<<<(Nn + 255) / 256, 256, 0, stream>>>(deg, dinv, Nn);
  edge_prep_kernel<<<(E + 255) / 256, 256, 0, stream>>>(ei, dinv, src32, dst32, wE, E);

  // layer 1: h1 = x @ W1 (WMMA), scatter, epilogue
  {
    dim3 grid((Nn + 63) / 64, HID / 16);
    wmma_gemm_f16_kernel<HID><<<grid, 128, 0, stream>>>(x16, w1t, h1, Nn);
  }
  {
    long long tot = (long long)E * (HID / 4);
    edge_scatter_kernel<HID><<<(unsigned)((tot + 255) / 256), 256, 0, stream>>>(
        h1, src32, dst32, wE, agg1, E);
  }
  finalize_relu_f16_kernel<<<(Nn * HID + 255) / 256, 256, 0, stream>>>(
      agg1, h1, dinv, b1, z16, Nn * HID, HID);

  // layer 2: h2 = z @ W2 (WMMA), scatter into d_out, epilogue
  {
    dim3 grid((Nn + 63) / 64, OUT_C / 16);
    wmma_gemm_f16_kernel<OUT_C><<<grid, 128, 0, stream>>>(z16, w2t, h2, Nn);
  }
  {
    long long tot = (long long)E * (OUT_C / 4);
    edge_scatter_kernel<OUT_C><<<(unsigned)((tot + 255) / 256), 256, 0, stream>>>(
        h2, src32, dst32, wE, out, E);
  }
  finalize_out_kernel<<<(Nn * OUT_C + 255) / 256, 256, 0, stream>>>(
      out, h2, dinv, b2, Nn * OUT_C, OUT_C);
}
